// CausalSelfAttention_29051158790829
// MI455X (gfx1250) — compile-verified, matrix operand order checked by reading, not run on hardware
//
#include <hip/hip_runtime.h>
#include <hip/hip_bf16.h>

typedef __bf16 bf16;
typedef __attribute__((ext_vector_type(16))) __bf16 v16bf;
typedef __attribute__((ext_vector_type(8)))  __bf16 v8bf;
typedef __attribute__((ext_vector_type(4)))  __bf16 v4bf;
typedef __attribute__((ext_vector_type(8)))  float  v8f;
typedef __attribute__((ext_vector_type(4)))  unsigned int u32x4;
typedef __attribute__((ext_vector_type(8)))  int i32x8;
typedef __attribute__((ext_vector_type(4)))  int i32x4;

#define HID   1024
#define SEQ   2048
#define BATCH 4
#define NHEAD 16
#define HDIM  64
#define MTOT  (BATCH*SEQ)   // 8192

#if defined(__has_builtin)
#  if __has_builtin(__builtin_amdgcn_tensor_load_to_lds)
#    define HAVE_TDM 1
#  endif
#endif
#ifndef HAVE_TDM
#  define HAVE_TDM 0
#endif

// ---------------------------------------------------------------------------
// WMMA helper: D = A(16x32 bf16) x B(32x16 bf16) + C(16x16 f32)
// ---------------------------------------------------------------------------
__device__ __forceinline__ v8f wmma_bf16(v16bf a, v16bf b, v8f c) {
  return __builtin_amdgcn_wmma_f32_16x16x32_bf16(false, a, false, b,
                                                 (short)0, c, false, false);
}

// Async global->LDS copy, 16 bytes per lane, tracked by ASYNCcnt.
__device__ __forceinline__ void async_b128(const bf16* lds_dst, const bf16* gsrc) {
  unsigned lds_off = (unsigned)(size_t)lds_dst;   // low 32 bits = LDS offset
  asm volatile("global_load_async_to_lds_b128 %0, %1, off"
               :: "v"(lds_off), "v"(gsrc) : "memory");
}
__device__ __forceinline__ void wait_async0() {
  asm volatile("s_wait_asynccnt 0x0" ::: "memory");
}

#if HAVE_TDM
// Arity shim: ROCm 7.2 (clang-22) = 5 args; therock (clang>=23) = 6 args.
__device__ __forceinline__ void tdm_issue(u32x4 g0, i32x8 g1) {
  i32x4 z4 = {0, 0, 0, 0};                             // groups 2/3 unused (2D)
#if defined(__clang_major__) && (__clang_major__ >= 23)
  i32x8 z8 = {0, 0, 0, 0, 0, 0, 0, 0};
  __builtin_amdgcn_tensor_load_to_lds(g0, g1, z4, z4, z8, 0);
#else
  __builtin_amdgcn_tensor_load_to_lds(g0, g1, z4, z4, 0);
#endif
}

// TDM: DMA one 2D bf16 tile (tile_rows x 32 elems) from a row-major tensor
// (row pitch 1024 elems) into LDS with 16B padding after each 64B row
// (reproduces the 40-element LDS pitch). One instruction per tile; TENSORcnt.
__device__ __forceinline__ void tdm_tile(const bf16* gsrc, const bf16* lds_dst,
                                         int tile_rows) {
  unsigned lds = (unsigned)(size_t)lds_dst;
  unsigned long long ga = (unsigned long long)(size_t)gsrc;
  u32x4 g0;
  g0[0] = 1u;                                          // count=1, user mode
  g0[1] = lds;                                         // lds_addr
  g0[2] = (unsigned)(ga & 0xFFFFFFFFu);                // global_addr[31:0]
  g0[3] = (unsigned)((ga >> 32) & 0x01FFFFFFu)         // global_addr[56:32]
        | (2u << 30);                                  // type=2 (image)
  i32x8 g1;
  g1[0] = (1 << 16)                                    // data_size = 2 bytes
        | (1 << 20)                                    // pad_enable
        | (3 << 22)                                    // pad_interval: 16 DWORDs (64B)
        | (3 << 25);                                   // pad_amount: 4 DWORDs (16B)
  g1[1] = (HID & 0xFFFF) << 16;                        // tensor_dim0[15:0]
  g1[2] = (HID >> 16) | ((tile_rows & 0xFFFF) << 16);  // dim0 hi | tensor_dim1 lo
  g1[3] = (tile_rows >> 16) | (32 << 16);              // dim1 hi | tile_dim0 = 32
  g1[4] = tile_rows & 0xFFFF;                          // tile_dim1 (tile_dim2 = 0)
  g1[5] = HID;                                         // tensor_dim0_stride[31:0]
  g1[6] = 0;
  g1[7] = 0;
  tdm_issue(g0, g1);
}
#endif

// Load one A/B fragment (16 rows x K=32, bf16) per CDNA5 wave32 layout:
// lane L: row = L%16, kb = (L>=16)*8; elems 0..7 = K kb..kb+7, 8..15 = K 16+kb..
__device__ __forceinline__ v16bf load_frag(const bf16* base, int stride, int lane) {
  int r  = lane & 15;
  int kb = (lane >> 4) << 3;
  const bf16* p = base + r * stride + kb;
  v8bf lo = *(const v8bf*)(p);
  v8bf hi = *(const v8bf*)(p + 16);
  v16bf o;
#pragma unroll
  for (int i = 0; i < 8; ++i) { o[i] = lo[i]; o[i + 8] = hi[i]; }
  return o;
}

// ---------------------------------------------------------------------------
// fp32 -> bf16 conversion (vector of 4)
// ---------------------------------------------------------------------------
__global__ void cvt_kernel(const float* __restrict__ in, bf16* __restrict__ out, int n4) {
  int i = blockIdx.x * blockDim.x + threadIdx.x;
  if (i < n4) {
    float4 v = ((const float4*)in)[i];
    v4bf o = { (bf16)v.x, (bf16)v.y, (bf16)v.z, (bf16)v.w };
    ((v4bf*)out)[i] = o;
  }
}

// ---------------------------------------------------------------------------
// GEMM: C[M,N] = A[M,K] @ W[N,K]^T + bias    (M=8192, N=K=1024, bf16 in, f32 acc)
// Block: 256 thr = 8 waves; tile 128(M) x 64(N); each wave 32x32 (2x2 WMMA accums)
// Staging: TDM tensor_load_to_lds (TENSORcnt) double-buffered; async fallback.
// ---------------------------------------------------------------------------
#define LDA 40   // 32 K elems + 8 pad (80B rows: 16B aligned; TDM pad reproduces)
#define ATILE (128 * LDA)
#define BTILE (64 * LDA)

__device__ __forceinline__ void gemm_core(const bf16* __restrict__ A,
                                          const bf16* __restrict__ W,
                                          int m0, int n0, int wy, int wx,
                                          int lane, bf16* As, bf16* Bs,
                                          v8f acc[2][2]) {
#if HAVE_TDM
  bool leader = (threadIdx.x < 32);    // wave 0 drives the DMA engine
  auto issue = [&](bf16* Asb, bf16* Bsb, int k0) {
    tdm_tile(A + (size_t)m0 * HID + k0, Asb, 128);
    tdm_tile(W + (size_t)n0 * HID + k0, Bsb, 64);
  };
  if (leader) issue(As, Bs, 0);
  int cur = 0;
  for (int k0 = 0; k0 < HID; k0 += 32) {
    bf16* Asb = As + cur * ATILE;
    bf16* Bsb = Bs + cur * BTILE;
    if (leader) {
      if (k0 + 32 < HID) {
        issue(As + (cur ^ 1) * ATILE, Bs + (cur ^ 1) * BTILE, k0 + 32);
        __builtin_amdgcn_s_wait_tensorcnt(2);   // current tile's 2 DMAs done
      } else {
        __builtin_amdgcn_s_wait_tensorcnt(0);
      }
    }
    __syncthreads();
#else
  int t  = threadIdx.x;
  int ra = t >> 1, hf = t & 1;     // A tile: 128 rows x 32 K, 32B/thread
  int rb = t >> 2, sg = t & 3;     // B tile:  64 rows x 32 K, 16B/thread
  auto issue = [&](bf16* Asb, bf16* Bsb, int k0) {
    const bf16* srcA = A + (size_t)(m0 + ra) * HID + k0 + hf * 16;
    async_b128(&Asb[ra * LDA + hf * 16],     srcA);
    async_b128(&Asb[ra * LDA + hf * 16 + 8], srcA + 8);
    const bf16* srcB = W + (size_t)(n0 + rb) * HID + k0 + sg * 8;
    async_b128(&Bsb[rb * LDA + sg * 8], srcB);
  };
  issue(As, Bs, 0);
  int cur = 0;
  for (int k0 = 0; k0 < HID; k0 += 32) {
    bf16* Asb = As + cur * ATILE;
    bf16* Bsb = Bs + cur * BTILE;
    if (k0 + 32 < HID) {
      issue(As + (cur ^ 1) * ATILE, Bs + (cur ^ 1) * BTILE, k0 + 32);
      asm volatile("s_wait_asynccnt 0x3" ::: "memory");
    } else {
      wait_async0();
    }
    __syncthreads();
#endif
    v16bf af0 = load_frag(&Asb[(wy * 32 +  0) * LDA], LDA, lane);
    v16bf af1 = load_frag(&Asb[(wy * 32 + 16) * LDA], LDA, lane);
    v16bf bf0 = load_frag(&Bsb[(wx * 32 +  0) * LDA], LDA, lane);
    v16bf bf1 = load_frag(&Bsb[(wx * 32 + 16) * LDA], LDA, lane);
    acc[0][0] = wmma_bf16(af0, bf0, acc[0][0]);
    acc[0][1] = wmma_bf16(af0, bf1, acc[0][1]);
    acc[1][0] = wmma_bf16(af1, bf0, acc[1][0]);
    acc[1][1] = wmma_bf16(af1, bf1, acc[1][1]);
    __syncthreads();
    cur ^= 1;
  }
}

__global__ __launch_bounds__(256)
void gemm_qkv_kernel(const bf16* __restrict__ A, const bf16* __restrict__ W,
                     const float* __restrict__ bias, bf16* __restrict__ out) {
  __shared__ bf16 As[2 * ATILE];
  __shared__ bf16 Bs[2 * BTILE];
  int bm = blockIdx.x >> 4, bn = blockIdx.x & 15;
  int m0 = bm * 128, n0 = bn * 64;
  int w = threadIdx.x >> 5, lane = threadIdx.x & 31;
  int wy = w >> 1, wx = w & 1;
  v8f acc[2][2] = {};
  gemm_core(A, W, m0, n0, wy, wx, lane, As, Bs, acc);
#pragma unroll
  for (int i = 0; i < 2; ++i)
#pragma unroll
    for (int j = 0; j < 2; ++j) {
      int n = n0 + wx * 32 + j * 16 + (lane & 15);
      float bn = bias[n];
      int h = n >> 6, d = n & 63;
#pragma unroll
      for (int v = 0; v < 8; ++v) {
        int m = m0 + wy * 32 + i * 16 + v + ((lane >> 4) << 3);
        int b = m >> 11, s = m & 2047;                  // m = b*SEQ + s
        out[(((size_t)b * NHEAD + h) * SEQ + s) * HDIM + d] = (bf16)(acc[i][j][v] + bn);
      }
    }
}

__global__ __launch_bounds__(256)
void gemm_out_kernel(const bf16* __restrict__ A, const bf16* __restrict__ W,
                     const float* __restrict__ bias, const float* __restrict__ X,
                     float* __restrict__ Y) {
  __shared__ bf16 As[2 * ATILE];
  __shared__ bf16 Bs[2 * BTILE];
  int bm = blockIdx.x >> 4, bn = blockIdx.x & 15;
  int m0 = bm * 128, n0 = bn * 64;
  int w = threadIdx.x >> 5, lane = threadIdx.x & 31;
  int wy = w >> 1, wx = w & 1;
  v8f acc[2][2] = {};
  gemm_core(A, W, m0, n0, wy, wx, lane, As, Bs, acc);
#pragma unroll
  for (int i = 0; i < 2; ++i)
#pragma unroll
    for (int j = 0; j < 2; ++j) {
      int n = n0 + wx * 32 + j * 16 + (lane & 15);
      float bn = bias[n];
#pragma unroll
      for (int v = 0; v < 8; ++v) {
        int m = m0 + wy * 32 + i * 16 + v + ((lane >> 4) << 3);
        Y[(size_t)m * HID + n] = acc[i][j][v] + bn + X[(size_t)m * HID + n];
      }
    }
}

// ---------------------------------------------------------------------------
// Flash attention (causal), bf16 QKV head-major [B,H,S,64], fp32 softmax/acc.
// Block = 256 thr = 8 waves; each wave owns 16 q-rows; 32-key steps via LDS.
// K tile staged by async global->LDS (ASYNCcnt); V transposed through VGPRs.
// ---------------------------------------------------------------------------
#define LDK 72   // K-tile row pitch (64 + 8 pad)
#define LDV 40   // transposed V / P row pitch (32 + 8 pad)

__global__ __launch_bounds__(256)
void attn_kernel(const bf16* __restrict__ Q, const bf16* __restrict__ K,
                 const bf16* __restrict__ V, bf16* __restrict__ Aout) {
  __shared__ bf16 Kt[32 * LDK];        // [key][d]
  __shared__ bf16 Vt[64 * LDV];        // [d][key]  (transposed)
  __shared__ bf16 Pt[8 * 16 * LDV];    // per-wave P, [qrow][key]
  int gid = blockIdx.x;
  int qb = gid & 15, h = (gid >> 4) & 15, b = gid >> 8;
  int w = threadIdx.x >> 5, lane = threadIdx.x & 31, t = threadIdx.x;
  int q0 = qb * 128 + w * 16;
  const bf16* Qh = Q + ((size_t)b * NHEAD + h) * SEQ * HDIM;
  const bf16* Kh = K + ((size_t)b * NHEAD + h) * SEQ * HDIM;
  const bf16* Vh = V + ((size_t)b * NHEAD + h) * SEQ * HDIM;

  // Q fragments stay resident: d 0..31 and 32..63
  v16bf qf0 = load_frag(Qh + (size_t)q0 * HDIM,      HDIM, lane);
  v16bf qf1 = load_frag(Qh + (size_t)q0 * HDIM + 32, HDIM, lane);

  v8f o[4] = {};
  float mrun[8], lrun[8];
#pragma unroll
  for (int v = 0; v < 8; ++v) { mrun[v] = -1e30f; lrun[v] = 0.f; }
  const float scale = 0.125f;          // 1/sqrt(64)
  int jmax = qb * 128 + 128;

  int r = t >> 3, sg = t & 7;          // 32 rows x 8 segs of 8 elems
  for (int kk = 0; kk < jmax; kk += 32) {
    // K tile via async pipe (16B/thread = whole 4KB tile)
    async_b128(&Kt[r * LDK + sg * 8], Kh + (size_t)(kk + r) * HDIM + sg * 8);
    // V tile: load to VGPRs, store transposed
    v8bf vv = *(const v8bf*)(Vh + (size_t)(kk + r) * HDIM + sg * 8);
#pragma unroll
    for (int j = 0; j < 8; ++j) Vt[(sg * 8 + j) * LDV + r] = vv[j];
    wait_async0();
    __syncthreads();

    if (kk <= q0 + 15) {               // wave-uniform causal skip (EXEC stays full)
      // S = Q @ K^T : two 16x16 score tiles (keys kk..+15, kk+16..+31)
      v8f s0 = {}, s1 = {};
      s0 = wmma_bf16(qf0, load_frag(&Kt[0],            LDK, lane), s0);
      s0 = wmma_bf16(qf1, load_frag(&Kt[32],           LDK, lane), s0);
      s1 = wmma_bf16(qf0, load_frag(&Kt[16 * LDK],     LDK, lane), s1);
      s1 = wmma_bf16(qf1, load_frag(&Kt[16 * LDK + 32],LDK, lane), s1);

      // online softmax in fp32; C-layout: lane%16 = col, lane/16 picks M-half
#pragma unroll
      for (int v = 0; v < 8; ++v) {
        int qrow = q0 + v + ((lane >> 4) << 3);
        int c0 = kk + (lane & 15), c1 = c0 + 16;
        float a0 = s0[v] * scale + (c0 > qrow ? -1e9f : 0.f);
        float a1 = s1[v] * scale + (c1 > qrow ? -1e9f : 0.f);
        float mx = fmaxf(a0, a1);
#pragma unroll
        for (int off = 1; off < 16; off <<= 1)
          mx = fmaxf(mx, __shfl_xor(mx, off, 32));
        float mnew  = fmaxf(mrun[v], mx);
        float alpha = __expf(mrun[v] - mnew);
        float p0 = __expf(a0 - mnew), p1 = __expf(a1 - mnew);
        float rs = p0 + p1;
#pragma unroll
        for (int off = 1; off < 16; off <<= 1)
          rs += __shfl_xor(rs, off, 32);
        lrun[v] = lrun[v] * alpha + rs;
        mrun[v] = mnew;
#pragma unroll
        for (int f = 0; f < 4; ++f) o[f][v] *= alpha;
        int rp = v + ((lane >> 4) << 3);
        Pt[(w * 16 + rp) * LDV + (lane & 15)]      = (bf16)p0;
        Pt[(w * 16 + rp) * LDV + 16 + (lane & 15)] = (bf16)p1;
      }
      asm volatile("s_wait_dscnt 0" ::: "memory");   // P store -> P frag read
      v16bf pf = load_frag(&Pt[w * 16 * LDV], LDV, lane);
#pragma unroll
      for (int f = 0; f < 4; ++f)                    // O += P @ V (d-tiles of 16)
        o[f] = wmma_bf16(pf, load_frag(&Vt[f * 16 * LDV], LDV, lane), o[f]);
    }
    __syncthreads();
  }

  // normalize and emit bf16 attn row-major [B*S, HID]
#pragma unroll
  for (int f = 0; f < 4; ++f)
#pragma unroll
    for (int v = 0; v < 8; ++v) {
      int mrow = v + ((lane >> 4) << 3);
      float val = o[f][v] / lrun[v];
      Aout[((size_t)(b * SEQ + q0 + mrow)) * HID + h * HDIM + f * 16 + (lane & 15)] = (bf16)val;
    }
}

// ---------------------------------------------------------------------------
// LayerNorm: one wave per 1024-wide row
// ---------------------------------------------------------------------------
__global__ __launch_bounds__(256)
void ln_kernel(const float* __restrict__ Y, const float* __restrict__ gamma,
               const float* __restrict__ beta, float* __restrict__ out) {
  int row = blockIdx.x * 8 + (threadIdx.x >> 5);
  int lane = threadIdx.x & 31;
  const float* p = Y + (size_t)row * HID;
  float s = 0.f, s2 = 0.f;
  for (int i = lane; i < HID; i += 32) { float v = p[i]; s += v; s2 += v * v; }
#pragma unroll
  for (int off = 16; off >= 1; off >>= 1) {
    s  += __shfl_xor(s,  off, 32);
    s2 += __shfl_xor(s2, off, 32);
  }
  float mean = s * (1.f / HID);
  float inv  = rsqrtf(s2 * (1.f / HID) - mean * mean + 1e-5f);
  float* q = out + (size_t)row * HID;
  for (int i = lane; i < HID; i += 32)
    q[i] = (p[i] - mean) * inv * gamma[i] + beta[i];
}

// ---------------------------------------------------------------------------
extern "C" void kernel_launch(void* const* d_in, const int* in_sizes, int n_in,
                              void* d_out, int out_size, void* d_ws, size_t ws_size,
                              hipStream_t stream) {
  const float* x  = (const float*)d_in[0];
  const float* Wq = (const float*)d_in[1];
  const float* bq = (const float*)d_in[2];
  const float* Wk = (const float*)d_in[3];
  const float* bk = (const float*)d_in[4];
  const float* Wv = (const float*)d_in[5];
  const float* bv = (const float*)d_in[6];
  const float* Wo = (const float*)d_in[7];
  const float* bo = (const float*)d_in[8];
  const float* gamma = (const float*)d_in[9];
  const float* beta  = (const float*)d_in[10];
  float* out = (float*)d_out;

  // workspace carve-up (~126 MB)
  bf16* xb  = (bf16*)d_ws;                         // 8192x1024
  bf16* wqb = xb  + (size_t)MTOT * HID;
  bf16* wkb = wqb + (size_t)HID * HID;
  bf16* wvb = wkb + (size_t)HID * HID;
  bf16* wob = wvb + (size_t)HID * HID;
  bf16* qb  = wob + (size_t)HID * HID;             // [B,H,S,64]
  bf16* kb  = qb  + (size_t)MTOT * HID;
  bf16* vb  = kb  + (size_t)MTOT * HID;
  bf16* ab  = vb  + (size_t)MTOT * HID;            // attn, [B*S, HID]
  float* y  = (float*)(ab + (size_t)MTOT * HID);   // pre-LN residual sum

  int nx4 = MTOT * HID / 4;
  cvt_kernel<<<nx4 / 256, 256, 0, stream>>>(x, xb, nx4);
  int nw4 = HID * HID / 4;
  cvt_kernel<<<nw4 / 256, 256, 0, stream>>>(Wq, wqb, nw4);
  cvt_kernel<<<nw4 / 256, 256, 0, stream>>>(Wk, wkb, nw4);
  cvt_kernel<<<nw4 / 256, 256, 0, stream>>>(Wv, wvb, nw4);
  cvt_kernel<<<nw4 / 256, 256, 0, stream>>>(Wo, wob, nw4);

  dim3 gg((MTOT / 128) * (HID / 64));              // 1024 blocks
  gemm_qkv_kernel<<<gg, 256, 0, stream>>>(xb, wqb, bq, qb);
  gemm_qkv_kernel<<<gg, 256, 0, stream>>>(xb, wkb, bk, kb);
  gemm_qkv_kernel<<<gg, 256, 0, stream>>>(xb, wvb, bv, vb);

  attn_kernel<<<BATCH * NHEAD * (SEQ / 128), 256, 0, stream>>>(qb, kb, vb, ab);

  gemm_out_kernel<<<gg, 256, 0, stream>>>(ab, wob, bo, x, y);
  ln_kernel<<<MTOT / 8, 256, 0, stream>>>(y, gamma, beta, out);
}
